// UserPreferenceAggregator_87497073754415
// MI455X (gfx1250) — matrix operand order repaired and checked by reading, not compiled, not run on hardware
//
#include <hip/hip_runtime.h>
#include <hip/hip_bf16.h>

typedef __attribute__((ext_vector_type(16))) _Float16 v16h;
typedef __attribute__((ext_vector_type(8)))  _Float16 v8h;
typedef __attribute__((ext_vector_type(8)))  float    v8f;

#define U_USERS 8192
#define N_ITEMS 100000
#define H_LEN   50
#define QD_DIM  128
#define D_DIM   256

// ---------------------------------------------------------------------------
// Pack helpers: hoist f32->f16 conversion (and B transpose) out of the GEMMs.
// ---------------------------------------------------------------------------
__global__ __launch_bounds__(256) void cvt_f16(const float* __restrict__ src,
                                               _Float16* __restrict__ dst,
                                               int n) {
  const int i = blockIdx.x * 256 + threadIdx.x;
  if (i < n) dst[i] = (_Float16)src[i];
}

// dst[n*K + k] = src[k*N + n]   (row-major [K,N] -> K-contiguous [N,K] f16)
__global__ __launch_bounds__(256) void transpose_f16(
    const float* __restrict__ src, _Float16* __restrict__ dst, int K, int N) {
  const int i = blockIdx.x * 256 + threadIdx.x;
  if (i < K * N) {
    const int n = i / K;
    const int k = i - n * K;
    dst[i] = (_Float16)src[(size_t)k * N + n];
  }
}

// ---------------------------------------------------------------------------
// WMMA GEMM, all-f16 operands: C[M,N] = A[M,K] x Bt[N,K]^T (+ bias terms)
//   A : f16 row-major [M,K]
//   Bt: f16 [N,K], K contiguous  -> B fragment = two 8-half K runs/lane
//   ROWSCALE: epilogue adds rowscale[m]*bias[n] instead of bias[n]
//   HALF_OUT: additionally emit f16 copy of C (feeds the next GEMM's A)
// One 16x16 tile per wave, 8 waves per 256-thread block.
// Fragment layouts per CDNA5 ISA 7.12.2 (wave32); fp32 accumulate.
// ---------------------------------------------------------------------------
template <bool ROWSCALE, bool HALF_OUT>
__global__ __launch_bounds__(256) void wmma_gemm_f16(
    const _Float16* __restrict__ A, const _Float16* __restrict__ Bt,
    const float* __restrict__ bias, const float* __restrict__ rowscale,
    float* __restrict__ C, _Float16* __restrict__ C16, int M, int N, int K) {
  const int lane = threadIdx.x & 31;
  const int wave = threadIdx.x >> 5;
  const int tiles_n = N >> 4;
  const int tile = blockIdx.x * 8 + wave;
  const int tm = (tile / tiles_n) << 4;
  const int tn = (tile % tiles_n) << 4;
  if (tm >= M) return;  // uniform per wave

  const int hi = lane >> 4;  // lane half selects K sub-range
  const int lo = lane & 15;
  const int n  = tn + lo;
  const _Float16* pa = A + (size_t)(tm + lo) * K + hi * 8;
  const _Float16* pb = Bt + (size_t)n * K + hi * 8;

  v8f c = {};
  for (int k0 = 0; k0 < K; k0 += 32) {
    // A 16x32 fragment: elems 0..7 -> K=k0+hi*8.., elems 8..15 -> +16
    const v8h a0 = *(const v8h*)(pa + k0);
    const v8h a1 = *(const v8h*)(pa + k0 + 16);
    // B 32x16 fragment: lane = col n, elems 0..7 -> K=k0+hi*8.., elems 8..15 -> +16
    const v8h b0 = *(const v8h*)(pb + k0);
    const v8h b1 = *(const v8h*)(pb + k0 + 16);
    v16h a, b;
#pragma unroll
    for (int e = 0; e < 8; ++e) {
      a[e] = a0[e]; a[e + 8] = a1[e];
      b[e] = b0[e]; b[e + 8] = b1[e];
    }
    c = __builtin_amdgcn_wmma_f32_16x16x32_f16(
        /*neg_a=*/false, a, /*neg_b=*/false, b,
        /*c_mod=*/(short)0, c, /*reuse_a=*/false, /*reuse_b=*/false);
  }

  // C/D: VGPR r holds row tm + hi*8 + r, col n
  const int mr = tm + hi * 8;
  const float bn = bias ? bias[n] : 0.0f;
#pragma unroll
  for (int r = 0; r < 8; ++r) {
    float v = c[r];
    if (ROWSCALE) v += rowscale[mr + r] * bn;
    else          v += bn;
    C[(size_t)(mr + r) * N + n] = v;
    if (HALF_OUT) C16[(size_t)(mr + r) * N + n] = (_Float16)v;
  }
}

// ---------------------------------------------------------------------------
// Attention pass: one block (256 threads = 8 waves) per user.
// Stage the user's 50x256 gathered features in LDS (51.2 KB), compute
// score_e = QK[u]·feat_e + P[u]·bk, faithful "sum-as-max" softmax, then
// weighted feature aggregate -> AGG16[u] (f16, feeds gemm3), WSUM[u].
// ---------------------------------------------------------------------------
__global__ __launch_bounds__(256) void attn_kernel(
    const float* __restrict__ QK, const float* __restrict__ P,
    const float* __restrict__ bk, const float* __restrict__ feats,
    const int* __restrict__ item_idx, _Float16* __restrict__ AGG16,
    float* __restrict__ WSUM) {
  __shared__ float sfeat[H_LEN * D_DIM];  // 51200 B
  __shared__ float sq[D_DIM];
  __shared__ float sscore[64];
  __shared__ float sweight[64];
  __shared__ float sred[8];

  const int u = blockIdx.x;
  const int t = threadIdx.x;
  const int lane = t & 31;
  const int wave = t >> 5;

  // load QK[u] row; block-reduce P[u]·bk
  sq[t] = QK[(size_t)u * D_DIM + t];
  float pb = P[(size_t)u * D_DIM + t] * bk[t];
#pragma unroll
  for (int off = 16; off; off >>= 1) pb += __shfl_xor(pb, off, 32);
  if (lane == 0) sred[wave] = pb;

  // gather this user's 50 item-feature rows into LDS (L2-hot)
  for (int i = t; i < H_LEN * D_DIM; i += 256) {
    const int e = i >> 8;
    const int d = i & 255;
    sfeat[i] = feats[(size_t)item_idx[u * H_LEN + e] * D_DIM + d];
  }
  __syncthreads();

  float qdotbk = 0.0f;
#pragma unroll
  for (int i = 0; i < 8; ++i) qdotbk += sred[i];

  // one wave per edge: 256-dot against sq
  for (int e = wave; e < H_LEN; e += 8) {
    float acc = 0.0f;
#pragma unroll
    for (int d0 = 0; d0 < D_DIM; d0 += 32)
      acc += sq[d0 + lane] * sfeat[e * D_DIM + d0 + lane];
#pragma unroll
    for (int off = 16; off; off >>= 1) acc += __shfl_xor(acc, off, 32);
    if (lane == 0) sscore[e] = acc + qdotbk;
  }
  __syncthreads();

  // faithful reproduction: "max" is actually the per-user SUM of scores
  if (wave == 0) {
    const float s0 = (lane      < H_LEN) ? sscore[lane]      : 0.0f;
    const float s1 = (lane + 32 < H_LEN) ? sscore[lane + 32] : 0.0f;
    float S = s0 + s1;
#pragma unroll
    for (int off = 16; off; off >>= 1) S += __shfl_xor(S, off, 32);
    const float w0 = (lane      < H_LEN) ? __expf(s0 - S) : 0.0f;
    const float w1 = (lane + 32 < H_LEN) ? __expf(s1 - S) : 0.0f;
    float sumexp = w0 + w1;
#pragma unroll
    for (int off = 16; off; off >>= 1) sumexp += __shfl_xor(sumexp, off, 32);
    const float inv = 1.0f / (sumexp + 1e-12f);
    if (lane < H_LEN)      sweight[lane]      = w0 * inv;
    if (lane + 32 < H_LEN) sweight[lane + 32] = w1 * inv;
    if (lane == 0) WSUM[u] = sumexp * inv;  // Σ attn_weights (feeds bv term)
  }
  __syncthreads();

  // weighted feature aggregate: thread t owns dim t (f32 accumulate)
  float acc = 0.0f;
#pragma unroll
  for (int e = 0; e < H_LEN; ++e) acc += sweight[e] * sfeat[e * D_DIM + t];
  AGG16[(size_t)u * D_DIM + t] = (_Float16)acc;
}

// ---------------------------------------------------------------------------
extern "C" void kernel_launch(void* const* d_in, const int* in_sizes, int n_in,
                              void* d_out, int out_size, void* d_ws,
                              size_t ws_size, hipStream_t stream) {
  (void)in_sizes; (void)n_in; (void)out_size; (void)ws_size;
  const float* uq    = (const float*)d_in[0];  // [U, QD]
  const float* feats = (const float*)d_in[1];  // [N, D]
  const float* Wq    = (const float*)d_in[2];  // [QD, D]
  const float* bq    = (const float*)d_in[3];  // [D]
  const float* Wk    = (const float*)d_in[4];  // [D, D]
  const float* bk    = (const float*)d_in[5];  // [D]
  const float* Wv    = (const float*)d_in[6];  // [D, D]
  const float* bv    = (const float*)d_in[7];  // [D]
  // d_in[8]: batch_user_indices == repeat(arange(U), H) -> implicit
  const int* item_idx = (const int*)d_in[9];   // [E]
  float* out = (float*)d_out;                  // [U, D]

  // workspace layout (all 256B-aligned)
  char* ws = (char*)d_ws;
  size_t off = 0;
  float*    P     = (float*)(ws + off);    off += (size_t)U_USERS * D_DIM * 4;
  float*    QK    = (float*)(ws + off);    off += (size_t)U_USERS * D_DIM * 4;
  float*    WSUM  = (float*)(ws + off);    off += (size_t)U_USERS * 4;
  _Float16* UQ16  = (_Float16*)(ws + off); off += (size_t)U_USERS * QD_DIM * 2;
  _Float16* P16   = (_Float16*)(ws + off); off += (size_t)U_USERS * D_DIM * 2;
  _Float16* AGG16 = (_Float16*)(ws + off); off += (size_t)U_USERS * D_DIM * 2;
  _Float16* Bq16  = (_Float16*)(ws + off); off += (size_t)QD_DIM * D_DIM * 2;
  _Float16* Bk16  = (_Float16*)(ws + off); off += (size_t)D_DIM * D_DIM * 2;
  _Float16* Bv16  = (_Float16*)(ws + off); off += (size_t)D_DIM * D_DIM * 2;

  const dim3 blk(256);

  // ---- pack: A operand uq -> f16; B operands -> f16 [N,K] (K-contiguous)
  const int nuq = U_USERS * QD_DIM;
  cvt_f16<<<dim3((nuq + 255) / 256), blk, 0, stream>>>(uq, UQ16, nuq);
  // Wq [QD,D] row-major -> Bq16 [D,QD]
  transpose_f16<<<dim3((QD_DIM * D_DIM + 255) / 256), blk, 0, stream>>>(
      Wq, Bq16, QD_DIM, D_DIM);
  // gemm2 uses B = Wk^T; Wk row-major [D,D] is already [N,K] for that: convert
  cvt_f16<<<dim3((D_DIM * D_DIM + 255) / 256), blk, 0, stream>>>(
      Wk, Bk16, D_DIM * D_DIM);
  // Wv [D,D] row-major -> Bv16 [D,D] transposed
  transpose_f16<<<dim3((D_DIM * D_DIM + 255) / 256), blk, 0, stream>>>(
      Wv, Bv16, D_DIM, D_DIM);

  const int tiles = (U_USERS / 16) * (D_DIM / 16);  // 8192 tiles
  const dim3 gemm_grid(tiles / 8);                  // 8 waves/block

  // 1) P = uq @ Wq + bq   (also emit f16 copy for gemm2's A)
  wmma_gemm_f16<false, true><<<gemm_grid, blk, 0, stream>>>(
      UQ16, Bq16, bq, nullptr, P, P16, U_USERS, D_DIM, QD_DIM);
  // 2) QK = P @ Wk^T
  wmma_gemm_f16<false, false><<<gemm_grid, blk, 0, stream>>>(
      P16, Bk16, nullptr, nullptr, QK, nullptr, U_USERS, D_DIM, D_DIM);
  // 3) attention + weighted feature aggregation (memory-bound gather pass)
  attn_kernel<<<dim3(U_USERS), blk, 0, stream>>>(
      QK, P, bk, feats, item_idx, AGG16, WSUM);
  // 4) out = AGG @ Wv + WSUM ⊗ bv
  wmma_gemm_f16<true, false><<<gemm_grid, blk, 0, stream>>>(
      AGG16, Bv16, bv, WSUM, out, nullptr, U_USERS, D_DIM, D_DIM);
}
